// Attention_77249281786063
// MI455X (gfx1250) — matrix operand order repaired and checked by reading, not compiled, not run on hardware
//
#include <hip/hip_runtime.h>
#include <stdint.h>

// ---------------- problem constants ----------------
#define BB    2
#define SS    2048
#define DIMM  4096
#define NH    32
#define NKV   8
#define HD    128
#define NREP  4
#define NTOK  (BB*SS)            // 4096 tokens

typedef __attribute__((ext_vector_type(16))) _Float16 v16h;
typedef __attribute__((ext_vector_type(8)))  float    v8f;

// -------- fragment loaders (match CDNA5 WMMA VGPR layouts) --------
// A 16x32 f16: lane L -> row M=L%16; elems: K = {half*8..+8} then {16+half*8..+8}
static __device__ __forceinline__ void ld_a_frag(v16h& f, const _Float16* p0, const _Float16* p1) {
  uint4* u = reinterpret_cast<uint4*>(&f);
  u[0] = *reinterpret_cast<const uint4*>(p0);
  u[1] = *reinterpret_cast<const uint4*>(p1);
}
// B 32x16 f16: lane L -> col N=L%16; elems: K = {half*8..+8} then {16+half*8..+8} (p = base + half*8)
static __device__ __forceinline__ void ld_b_frag(v16h& f, const _Float16* p) {
  uint4* u = reinterpret_cast<uint4*>(&f);
  const uint4* q = reinterpret_cast<const uint4*>(p);
  u[0] = q[0];
  u[1] = q[2];
}

// ---------------- conversion kernels ----------------
__global__ void f2h_copy(const float* __restrict__ src, _Float16* __restrict__ dst, int n) {
  int i = blockIdx.x * blockDim.x + threadIdx.x;
  if (i < n) dst[i] = (_Float16)src[i];
}

// W[K,N] f32  ->  Wt[N,K] f16, tiled 32x32 through LDS for coalescing
__global__ __launch_bounds__(256) void f2h_transpose(const float* __restrict__ W,
                                                     _Float16* __restrict__ Wt,
                                                     int K, int N) {
  __shared__ _Float16 t[32][33];
  int n0 = blockIdx.x * 32, k0 = blockIdx.y * 32;
  #pragma unroll
  for (int i = 0; i < 4; ++i)
    t[threadIdx.y + i * 8][threadIdx.x] =
        (_Float16)W[(size_t)(k0 + threadIdx.y + i * 8) * N + n0 + threadIdx.x];
  __syncthreads();
  #pragma unroll
  for (int i = 0; i < 4; ++i)
    Wt[(size_t)(n0 + threadIdx.y + i * 8) * K + k0 + threadIdx.x] =
        t[threadIdx.x][threadIdx.y + i * 8];
}

// ---------------- generic WMMA GEMM:  C[M,N] = A[M,K] * Wt[N,K]^T ----------------
// 8 waves per block; wave w computes a 32(M) x 64(N) tile (2 A-frags share each
// B-frag -> 1.5 vmem issues per WMMA). Block covers 256M x 64N.
__global__ __launch_bounds__(256) void gemm_wmma(const _Float16* __restrict__ A,
                                                 const _Float16* __restrict__ Wt,
                                                 void* __restrict__ C,
                                                 int M, int N, int K, int c_is_f32) {
  int lane = threadIdx.x & 31;
  int wv   = threadIdx.x >> 5;
  int half = lane >> 4;
  int l16  = lane & 15;
  int m0 = blockIdx.y * 256 + wv * 32;
  int n0 = blockIdx.x * 64;

  const _Float16* arow0 = A + (size_t)(m0 + l16) * K;
  const _Float16* arow1 = A + (size_t)(m0 + 16 + l16) * K;
  v8f acc[2][4] = {};

  for (int kk = 0; kk < K; kk += 32) {
    v16h a0, a1;
    ld_a_frag(a0, arow0 + kk + half * 8, arow0 + kk + 16 + half * 8);
    ld_a_frag(a1, arow1 + kk + half * 8, arow1 + kk + 16 + half * 8);
    #pragma unroll
    for (int j = 0; j < 4; ++j) {
      v16h b;
      ld_b_frag(b, Wt + (size_t)(n0 + j * 16 + l16) * K + kk + half * 8);
      acc[0][j] = __builtin_amdgcn_wmma_f32_16x16x32_f16(false, a0, false, b,
                                                         (short)0, acc[0][j], false, false);
      acc[1][j] = __builtin_amdgcn_wmma_f32_16x16x32_f16(false, a1, false, b,
                                                         (short)0, acc[1][j], false, false);
    }
  }

  // C layout: VGPR r -> row m0 + i*16 + r + 8*half, col n0 + j*16 + l16
  if (c_is_f32) {
    float* Cf = (float*)C;
    #pragma unroll
    for (int i = 0; i < 2; ++i)
      #pragma unroll
      for (int j = 0; j < 4; ++j)
        #pragma unroll
        for (int r = 0; r < 8; ++r)
          Cf[(size_t)(m0 + i * 16 + 8 * half + r) * N + n0 + j * 16 + l16] = acc[i][j][r];
  } else {
    _Float16* Ch = (_Float16*)C;
    #pragma unroll
    for (int i = 0; i < 2; ++i)
      #pragma unroll
      for (int j = 0; j < 4; ++j)
        #pragma unroll
        for (int r = 0; r < 8; ++r)
          Ch[(size_t)(m0 + i * 16 + 8 * half + r) * N + n0 + j * 16 + l16] =
              (_Float16)acc[i][j][r];
  }
}

// ---------------- RoPE + re-layout ----------------
// q_lin [tok, H*128] -> qh [B,H,S,128] (rotary applied)
__global__ void rope_q(const _Float16* __restrict__ ql, const float* __restrict__ fc,
                       const float* __restrict__ fs, _Float16* __restrict__ qh) {
  int idx = blockIdx.x * blockDim.x + threadIdx.x;        // B*S*H*64
  if (idx >= BB * SS * NH * 64) return;
  int d2 = idx & 63;
  int t  = idx >> 6;
  int h  = t & 31;
  int u  = t >> 5;            // b*S + s
  int s  = u & (SS - 1);
  int b  = u >> 11;
  float c  = fc[s * 64 + d2];
  float sn = fs[s * 64 + d2];
  size_t in = (size_t)u * DIMM + h * HD + 2 * d2;
  float xr = (float)ql[in], xi = (float)ql[in + 1];
  size_t out = (((size_t)(b * NH + h) * SS) + s) * HD + 2 * d2;
  qh[out]     = (_Float16)(xr * c - xi * sn);
  qh[out + 1] = (_Float16)(xr * sn + xi * c);
}

// k_lin [tok, 1024] -> kh [B,Hkv,S,128] (rotary applied)
__global__ void rope_k(const _Float16* __restrict__ kl, const float* __restrict__ fc,
                       const float* __restrict__ fs, _Float16* __restrict__ kh) {
  int idx = blockIdx.x * blockDim.x + threadIdx.x;        // B*S*Hkv*64
  if (idx >= BB * SS * NKV * 64) return;
  int d2 = idx & 63;
  int t  = idx >> 6;
  int h  = t & 7;
  int u  = t >> 3;            // b*S + s
  int s  = u & (SS - 1);
  int b  = u >> 11;
  float c  = fc[s * 64 + d2];
  float sn = fs[s * 64 + d2];
  size_t in = (size_t)u * (NKV * HD) + h * HD + 2 * d2;
  float xr = (float)kl[in], xi = (float)kl[in + 1];
  size_t out = (((size_t)(b * NKV + h) * SS) + s) * HD + 2 * d2;
  kh[out]     = (_Float16)(xr * c - xi * sn);
  kh[out + 1] = (_Float16)(xr * sn + xi * c);
}

// v_lin [tok, Hkv*128] -> vt [B,Hkv,128,S]  (tiled transpose per (b,h))
__global__ __launch_bounds__(256) void transp_v(const _Float16* __restrict__ vl,
                                                _Float16* __restrict__ vt) {
  __shared__ _Float16 t[32][33];
  int bh = blockIdx.z;
  int b = bh >> 3, h = bh & 7;
  const _Float16* src = vl + (size_t)b * SS * (NKV * HD) + h * HD;
  _Float16* dst = vt + (size_t)bh * HD * SS;
  int s0 = blockIdx.x * 32, d0 = blockIdx.y * 32;
  #pragma unroll
  for (int i = 0; i < 4; ++i)
    t[threadIdx.y + i * 8][threadIdx.x] =
        src[(size_t)(s0 + threadIdx.y + i * 8) * (NKV * HD) + d0 + threadIdx.x];
  __syncthreads();
  #pragma unroll
  for (int i = 0; i < 4; ++i)
    dst[(size_t)(d0 + threadIdx.y + i * 8) * SS + s0 + threadIdx.x] =
        t[threadIdx.x][threadIdx.y + i * 8];
}

// ---------------- flash attention (one wave per 16 query rows) ----------------
__global__ __launch_bounds__(32) void attn_kernel(const _Float16* __restrict__ qh,
                                                  const _Float16* __restrict__ kh,
                                                  const _Float16* __restrict__ vt,
                                                  _Float16* __restrict__ ao) {
  const int lane = threadIdx.x;
  const int half = lane >> 4;
  const int l16  = lane & 15;
  const int qt = blockIdx.x, h = blockIdx.y, b = blockIdx.z;
  const int hk = h >> 2;                        // GQA: 4 q-heads per kv-head
  const float scale = 0.08838834764831845f;     // 1/sqrt(128)

  const _Float16* qbase = qh + (((size_t)(b * NH + h) * SS) + qt * 16) * HD;
  const _Float16* kbase = kh + ((size_t)(b * NKV + hk) * SS) * HD;
  const _Float16* vbase = vt + ((size_t)(b * NKV + hk) * HD) * SS;

  // Q fragments for the whole 16x128 tile (4 chunks of K=32)
  v16h qf[4];
  #pragma unroll
  for (int kk = 0; kk < 4; ++kk) {
    const _Float16* p = qbase + (size_t)l16 * HD + kk * 32;
    ld_a_frag(qf[kk], p + half * 8, p + 16 + half * 8);
  }

  v8f acc[8] = {};
  float m[8], l[8];
  #pragma unroll
  for (int r = 0; r < 8; ++r) { m[r] = -3.0e38f; l[r] = 0.0f; }

  __shared__ __align__(16) _Float16 pshm[16 * 32];

  for (int kb = 0; kb < SS; kb += 32) {
    // scores: two 16x16 tiles (keys kb..kb+15 and kb+16..kb+31)
    v8f s0 = {}, s1 = {};
    const _Float16* k0p = kbase + (size_t)(kb + l16) * HD;
    const _Float16* k1p = kbase + (size_t)(kb + 16 + l16) * HD;
    #pragma unroll
    for (int kk = 0; kk < 4; ++kk) {
      v16h bf;
      ld_b_frag(bf, k0p + kk * 32 + half * 8);
      s0 = __builtin_amdgcn_wmma_f32_16x16x32_f16(false, qf[kk], false, bf,
                                                  (short)0, s0, false, false);
    }
    #pragma unroll
    for (int kk = 0; kk < 4; ++kk) {
      v16h bf;
      ld_b_frag(bf, k1p + kk * 32 + half * 8);
      s1 = __builtin_amdgcn_wmma_f32_16x16x32_f16(false, qf[kk], false, bf,
                                                  (short)0, s1, false, false);
    }

    // online softmax per row (row = r + 8*half; 16 lanes of a half share a row)
    float alpha[8];
    #pragma unroll
    for (int r = 0; r < 8; ++r) {
      float v0 = s0[r] * scale;
      float v1 = s1[r] * scale;
      float mx = fmaxf(v0, v1);
      #pragma unroll
      for (int off = 1; off < 16; off <<= 1) mx = fmaxf(mx, __shfl_xor(mx, off, 32));
      float mnew = fmaxf(m[r], mx);
      float a = __expf(m[r] - mnew);
      float p0 = __expf(v0 - mnew);
      float p1 = __expf(v1 - mnew);
      float rs = p0 + p1;
      #pragma unroll
      for (int off = 1; off < 16; off <<= 1) rs += __shfl_xor(rs, off, 32);
      l[r] = l[r] * a + rs;
      m[r] = mnew;
      alpha[r] = a;
      int row = r + 8 * half;
      pshm[row * 32 + l16]      = (_Float16)p0;
      pshm[row * 32 + 16 + l16] = (_Float16)p1;
    }

    // rescale accumulators
    #pragma unroll
    for (int j = 0; j < 8; ++j)
      #pragma unroll
      for (int r = 0; r < 8; ++r) acc[j][r] *= alpha[r];

    // P as A-fragment via LDS round trip (C-layout -> A-layout reshape)
    v16h pf;
    {
      const _Float16* pb = &pshm[l16 * 32];
      ld_a_frag(pf, pb + half * 8, pb + 16 + half * 8);
    }

    // acc += P(16x32) x V(32x128), contiguous key runs in vt
    #pragma unroll
    for (int j = 0; j < 8; ++j) {
      v16h vf;
      ld_b_frag(vf, vbase + (size_t)(j * 16 + l16) * SS + kb + half * 8);
      acc[j] = __builtin_amdgcn_wmma_f32_16x16x32_f16(false, pf, false, vf,
                                                      (short)0, acc[j], false, false);
    }
  }

  // epilogue: divide by l, write [tok, H*128]
  float rl[8];
  #pragma unroll
  for (int r = 0; r < 8; ++r) rl[r] = 1.0f / l[r];
  _Float16* aobase = ao + ((size_t)(b * SS + qt * 16)) * DIMM + h * HD;
  #pragma unroll
  for (int j = 0; j < 8; ++j)
    #pragma unroll
    for (int r = 0; r < 8; ++r)
      aobase[(size_t)(r + 8 * half) * DIMM + j * 16 + l16] = (_Float16)(acc[j][r] * rl[r]);
}

// ---------------- workspace layout (elements of _Float16) ----------------
static constexpr size_t OFF_XB   = 0;          // 16777216  x f16 [tok,4096]
static constexpr size_t OFF_WQT  = 16777216;   // 16777216  wq^T [4096,4096]
static constexpr size_t OFF_WKT  = 33554432;   //  4194304  wk^T [1024,4096]
static constexpr size_t OFF_WVT  = 37748736;   //  4194304  wv^T [1024,4096]
static constexpr size_t OFF_WOT  = 41943040;   // 16777216  wo^T [4096,4096]
static constexpr size_t OFF_QLIN = 58720256;   // 16777216  q pre-rope (reused as AO)
static constexpr size_t OFF_KLIN = 75497472;   //  4194304
static constexpr size_t OFF_VLIN = 79691776;   //  4194304
static constexpr size_t OFF_QH   = 83886080;   // 16777216  [B,H,S,128]
static constexpr size_t OFF_KH   = 100663296;  //  4194304  [B,Hkv,S,128]
static constexpr size_t OFF_VT   = 104857600;  //  4194304  [B,Hkv,128,S]
static constexpr size_t OFF_AO   = OFF_QLIN;   // alias: q_lin dead after rope
// total = 109,051,904 elems = ~218 MB

extern "C" void kernel_launch(void* const* d_in, const int* in_sizes, int n_in,
                              void* d_out, int out_size, void* d_ws, size_t ws_size,
                              hipStream_t stream) {
  const float* x  = (const float*)d_in[0];
  const float* fc = (const float*)d_in[1];
  const float* fs = (const float*)d_in[2];
  const float* wq = (const float*)d_in[3];
  const float* wk = (const float*)d_in[4];
  const float* wv = (const float*)d_in[5];
  const float* wo = (const float*)d_in[6];
  float* out = (float*)d_out;
  _Float16* ws = (_Float16*)d_ws;

  // 1) conversions / weight transposes
  {
    int n = NTOK * DIMM;
    f2h_copy<<<n / 256, 256, 0, stream>>>(x, ws + OFF_XB, n);
  }
  f2h_transpose<<<dim3(DIMM / 32, DIMM / 32), dim3(32, 8), 0, stream>>>(wq, ws + OFF_WQT, DIMM, DIMM);
  f2h_transpose<<<dim3(1024 / 32, DIMM / 32), dim3(32, 8), 0, stream>>>(wk, ws + OFF_WKT, DIMM, 1024);
  f2h_transpose<<<dim3(1024 / 32, DIMM / 32), dim3(32, 8), 0, stream>>>(wv, ws + OFF_WVT, DIMM, 1024);
  f2h_transpose<<<dim3(DIMM / 32, DIMM / 32), dim3(32, 8), 0, stream>>>(wo, ws + OFF_WOT, DIMM, DIMM);

  // 2) q/k/v projections (WMMA)
  gemm_wmma<<<dim3(DIMM / 64, NTOK / 256), 256, 0, stream>>>(
      ws + OFF_XB, ws + OFF_WQT, (void*)(ws + OFF_QLIN), NTOK, DIMM, DIMM, 0);
  gemm_wmma<<<dim3(1024 / 64, NTOK / 256), 256, 0, stream>>>(
      ws + OFF_XB, ws + OFF_WKT, (void*)(ws + OFF_KLIN), NTOK, 1024, DIMM, 0);
  gemm_wmma<<<dim3(1024 / 64, NTOK / 256), 256, 0, stream>>>(
      ws + OFF_XB, ws + OFF_WVT, (void*)(ws + OFF_VLIN), NTOK, 1024, DIMM, 0);

  // 3) RoPE + layouts
  rope_q<<<(BB * SS * NH * 64) / 256, 256, 0, stream>>>(ws + OFF_QLIN, fc, fs, ws + OFF_QH);
  rope_k<<<(BB * SS * NKV * 64) / 256, 256, 0, stream>>>(ws + OFF_KLIN, fc, fs, ws + OFF_KH);
  transp_v<<<dim3(SS / 32, HD / 32, BB * NKV), dim3(32, 8), 0, stream>>>(ws + OFF_VLIN, ws + OFF_VT);

  // 4) attention (non-causal, flash-style)
  attn_kernel<<<dim3(SS / 16, NH, BB), 32, 0, stream>>>(
      ws + OFF_QH, ws + OFF_KH, ws + OFF_VT, ws + OFF_AO);

  // 5) output projection, f32 result straight to d_out
  gemm_wmma<<<dim3(DIMM / 64, NTOK / 256), 256, 0, stream>>>(
      ws + OFF_AO, ws + OFF_WOT, d_out, NTOK, DIMM, DIMM, 1);
  (void)in_sizes; (void)n_in; (void)out_size; (void)ws_size; (void)out;
}
